// TalkingHeadAttention_56470230008343
// MI455X (gfx1250) — compile-verified, matrix operand order checked by reading, not run on hardware
//
#include <hip/hip_runtime.h>

typedef __attribute__((ext_vector_type(16))) _Float16 v16h;
typedef __attribute__((ext_vector_type(8)))  float    v8f;

#define Bb 2
#define Nn 2048
#define Cc 1024
#define Hh 16
#define Dd 64
#define NEGBIG (-1e9f)

static __device__ __forceinline__ v8f wmma_f16(v16h a, v16h b, v8f c) {
  // D = A(16x32 f16) x B(32x16 f16) + C(16x16 f32)
  return __builtin_amdgcn_wmma_f32_16x16x32_f16(false, a, false, b, (short)0, c, false, false);
}

// Async global->LDS copy: per lane, 16 bytes from gptr to LDS byte offset
// lds_off (wave-relative).  Tracked with ASYNCcnt (CDNA5 ISA).
static __device__ __forceinline__ void async_copy16(unsigned lds_off, const void* gptr) {
  asm volatile("global_load_async_to_lds_b128 %0, %1, off"
               :: "v"(lds_off), "v"(gptr) : "memory");
}
// Wait until this wave's outstanding async copies <= N (in-order completion).
#define WAIT_ASYNC(n) asm volatile("s_wait_asynccnt " #n ::: "memory")

// LDS byte offset of a __shared__ pointer: flat LDS aperture stores the
// wave-relative offset in addr[31:0] (ISA 10.2 aperture mapping).
static __device__ __forceinline__ unsigned lds_off_of(const void* p) {
  return (unsigned)(size_t)p;
}

// ---------------------------------------------------------------------------
// Prep kernel A: f32 -> f16 elementwise (for x).
// ---------------------------------------------------------------------------
__global__ __launch_bounds__(256) void cvt_f16(const float* __restrict__ in,
                                               _Float16* __restrict__ out, int n4) {
  int i = (blockIdx.x * 256 + threadIdx.x);
  if (i < n4) {
    float4 v = *(const float4*)(in + i * 4);
    _Float16* o = out + i * 4;
    o[0] = (_Float16)v.x; o[1] = (_Float16)v.y;
    o[2] = (_Float16)v.z; o[3] = (_Float16)v.w;
  }
}

// ---------------------------------------------------------------------------
// Prep kernel B: transpose-convert W[rows][cols] f32 -> WT[cols][rows] f16
// (LDS-tiled so both global accesses are coalesced).
// ---------------------------------------------------------------------------
__global__ __launch_bounds__(256) void transpose_f16(const float* __restrict__ W,
                                                     _Float16* __restrict__ WT,
                                                     int rows, int cols) {
  __shared__ float t[32][33];
  const int c0 = blockIdx.x * 32, r0 = blockIdx.y * 32;
  const int tx = threadIdx.x & 31, ty = threadIdx.x >> 5;   // ty in 0..7
  for (int i = 0; i < 32; i += 8)
    t[ty + i][tx] = W[(r0 + ty + i) * cols + c0 + tx];
  __syncthreads();
  for (int i = 0; i < 32; i += 8)
    WT[(c0 + ty + i) * rows + r0 + tx] = (_Float16)t[tx][ty + i];
}

// ---------------------------------------------------------------------------
// Kernel 1: QKV projection GEMM.  xh[4096,1024] f16 @ WqkvT[3072,1024] f16.
// Block tile 64x128, 256 thr = 8 waves (4 row x 2 col), wave owns 16x64.
// Tiles staged global->LDS with async b128 copies, double buffered:
// issue next tile (3 async instr/wave), s_wait_asynccnt 3 -> current tile
// complete (in-order), barrier, 4 WMMAs, barrier.
// Epilogue scatters q (x0.125) / k as [b,h,n,d], v transposed as [b,h,d,n].
// ---------------------------------------------------------------------------
__global__ __launch_bounds__(256) void qkv_gemm(
    const _Float16* __restrict__ xh, const _Float16* __restrict__ WT,
    const float* __restrict__ bias,
    _Float16* __restrict__ qb, _Float16* __restrict__ kb,
    _Float16* __restrict__ vT)
{
  __shared__ _Float16 As[2][64 * 32];
  __shared__ _Float16 Bs[2][128 * 32];
  const int tid = threadIdx.x;
  const int l  = tid & 31;
  const int w  = tid >> 5;
  const int wr = w & 3, wc = w >> 2;
  const int lg = l & 15, lh = l >> 4;
  const int row0 = blockIdx.y * 64;
  const int col0 = blockIdx.x * 128;

  const unsigned asB = lds_off_of(&As[0][0]);
  const unsigned bsB = lds_off_of(&Bs[0][0]);
  // Per-lane 16B chunk assignment (A: 256 chunks, B: 512 chunks per tile)
  const int ar = tid >> 2, ap = tid & 3;
  const unsigned aLds = (unsigned)((ar * 32 + ap * 8) * 2);
  const _Float16* aG = xh + (row0 + ar) * 1024 + ap * 8;
  const int bc = tid >> 2, bp = tid & 3;
  const unsigned bLds0 = (unsigned)((bc * 32 + bp * 8) * 2);
  const unsigned bLds1 = (unsigned)(((bc + 64) * 32 + bp * 8) * 2);
  const _Float16* bG0 = WT + (col0 + bc) * 1024 + bp * 8;
  const _Float16* bG1 = WT + (col0 + bc + 64) * 1024 + bp * 8;

  // stage tile 0 into buffer 0
  async_copy16(asB + aLds, aG);
  async_copy16(bsB + bLds0, bG0);
  async_copy16(bsB + bLds1, bG1);

  v8f acc[4] = {};
  for (int it = 0; it < 32; ++it) {
    const int cur = it & 1;
    if (it + 1 < 32) {
      const int nk = (it + 1) * 32;
      const unsigned ao = asB + (unsigned)((cur ^ 1) * 64 * 32 * 2);
      const unsigned bo = bsB + (unsigned)((cur ^ 1) * 128 * 32 * 2);
      async_copy16(ao + aLds, aG + nk);
      async_copy16(bo + bLds0, bG0 + nk);
      async_copy16(bo + bLds1, bG1 + nk);
      WAIT_ASYNC(0x3);           // previous tile's 3 copies done
    } else {
      WAIT_ASYNC(0x0);
    }
    __syncthreads();
    const _Float16* Ac = As[cur];
    const _Float16* Bc = Bs[cur];
    v16h a = *(const v16h*)&Ac[(wr * 16 + lg) * 32 + lh * 16];
    for (int c = 0; c < 4; ++c) {
      v16h bf = *(const v16h*)&Bc[(wc * 64 + c * 16 + lg) * 32 + lh * 16];
      acc[c] = wmma_f16(a, bf, acc[c]);
    }
    __syncthreads();
  }
  for (int c = 0; c < 4; ++c)
    for (int j = 0; j < 8; ++j) {
      int gr = row0 + wr * 16 + j + 8 * lh;
      int gc = col0 + wc * 64 + c * 16 + lg;
      float v = acc[c][j] + bias[gc];
      int part = gc >> 10, hd = (gc >> 6) & 15, d = gc & 63;
      int b = gr >> 11, n = gr & 2047;
      if (part == 0)      qb[((b * 16 + hd) * 2048 + n) * 64 + d] = (_Float16)(v * 0.125f);
      else if (part == 1) kb[((b * 16 + hd) * 2048 + n) * 64 + d] = (_Float16)v;
      else                vT[((b * 16 + hd) * 64 + d) * 2048 + n] = (_Float16)v;
    }
}

// ---------------------------------------------------------------------------
// Kernel 2: softmax stats of the pre-mixed logits (flash pass A).
// 16 waves = 16 heads per (b, 16-row) block; QK^T per head via WMMA (K=64),
// tiles through LDS, head-mix with Wl, online (max,sum) per row via 16-lane
// xor-shuffle reductions.
// ---------------------------------------------------------------------------
__global__ __launch_bounds__(512) void attn_stats(
    const _Float16* __restrict__ qb, const _Float16* __restrict__ kb,
    const unsigned char* __restrict__ mask,
    const float* __restrict__ Wl, const float* __restrict__ bl,
    float* __restrict__ Mst, float* __restrict__ Sst)
{
  __shared__ float sS[16 * 2 * 256];
  const int tid = threadIdx.x, l = tid & 31, h = tid >> 5;
  const int lg = l & 15, lh = l >> 4;
  const int b = blockIdx.x >> 7, n0 = (blockIdx.x & 127) * 16;

  const _Float16* qp = qb + ((b * 16 + h) * 2048 + n0 + lg) * 64 + lh * 16;
  v16h aq0 = *(const v16h*)qp;
  v16h aq1 = *(const v16h*)(qp + 32);

  float wl[16];
  for (int g = 0; g < 16; ++g) wl[g] = Wl[g * 16 + h];
  const float blh = bl[h];

  float rM[8], rS[8];
  for (int j = 0; j < 8; ++j) { rM[j] = -3.0e38f; rS[j] = 0.f; }

  for (int m0 = 0; m0 < 2048; m0 += 32) {
    for (int t = 0; t < 2; ++t) {
      const _Float16* kp = kb + ((b * 16 + h) * 2048 + m0 + t * 16 + lg) * 64 + lh * 16;
      v8f s = {};
      s = wmma_f16(aq0, *(const v16h*)kp, s);
      s = wmma_f16(aq1, *(const v16h*)(kp + 32), s);
      float* o = &sS[(h * 2 + t) * 256 + l * 8];
      for (int j = 0; j < 8; ++j) o[j] = s[j];
    }
    __syncthreads();
    float madd0 = mask[b * 2048 + m0 + lg]      ? 0.f : NEGBIG;
    float madd1 = mask[b * 2048 + m0 + 16 + lg] ? 0.f : NEGBIG;
    for (int j = 0; j < 8; ++j) {
      float L0 = blh + madd0, L1 = blh + madd1;
      for (int g = 0; g < 16; ++g) {
        L0 += wl[g] * sS[(g * 2 + 0) * 256 + l * 8 + j];
        L1 += wl[g] * sS[(g * 2 + 1) * 256 + l * 8 + j];
      }
      float mx = fmaxf(L0, L1);
      for (int d = 1; d < 16; d <<= 1) mx = fmaxf(mx, __shfl_xor(mx, d));
      float mnew = fmaxf(rM[j], mx);
      float e = __expf(L0 - mnew) + __expf(L1 - mnew);
      for (int d = 1; d < 16; d <<= 1) e += __shfl_xor(e, d);
      rS[j] = rS[j] * __expf(rM[j] - mnew) + e;
      rM[j] = mnew;
    }
    __syncthreads();
  }
  if (lg == 0)
    for (int j = 0; j < 8; ++j) {
      int r = n0 + j + 8 * lh;
      Mst[(b * 16 + h) * 2048 + r] = rM[j];
      Sst[(b * 16 + h) * 2048 + r] = rS[j];
    }
}

// ---------------------------------------------------------------------------
// Kernel 3: flash pass B.  Recompute logits, normalize with stored stats,
// P -> LDS (f16, row-major), post-mix with Ww straight into an A-fragment,
// 4 PV WMMAs (K=32) per m-step against transposed V.
// ---------------------------------------------------------------------------
__global__ __launch_bounds__(512) void attn_out_k(
    const _Float16* __restrict__ qb, const _Float16* __restrict__ kb,
    const _Float16* __restrict__ vT, const unsigned char* __restrict__ mask,
    const float* __restrict__ Wl, const float* __restrict__ bl,
    const float* __restrict__ Ww, const float* __restrict__ bw,
    const float* __restrict__ Mst, const float* __restrict__ Sst,
    _Float16* __restrict__ aout)
{
  __shared__ float    sS[16 * 2 * 256];
  __shared__ _Float16 sP[16 * 16 * 32];
  const int tid = threadIdx.x, l = tid & 31, h = tid >> 5;
  const int lg = l & 15, lh = l >> 4;
  const int b = blockIdx.x >> 7, n0 = (blockIdx.x & 127) * 16;

  const _Float16* qp = qb + ((b * 16 + h) * 2048 + n0 + lg) * 64 + lh * 16;
  v16h aq0 = *(const v16h*)qp;
  v16h aq1 = *(const v16h*)(qp + 32);

  float wl[16], ww[16];
  for (int g = 0; g < 16; ++g) { wl[g] = Wl[g * 16 + h]; ww[g] = Ww[g * 16 + h]; }
  const float blh = bl[h], bwh = bw[h];

  float Mrow[8], iS[8];
  for (int j = 0; j < 8; ++j) {
    int r = n0 + j + 8 * lh;
    Mrow[j] = Mst[(b * 16 + h) * 2048 + r];
    iS[j]   = 1.f / fmaxf(Sst[(b * 16 + h) * 2048 + r], 1e-30f);
  }

  v8f o[4] = {};
  for (int m0 = 0; m0 < 2048; m0 += 32) {
    for (int t = 0; t < 2; ++t) {
      const _Float16* kp = kb + ((b * 16 + h) * 2048 + m0 + t * 16 + lg) * 64 + lh * 16;
      v8f s = {};
      s = wmma_f16(aq0, *(const v16h*)kp, s);
      s = wmma_f16(aq1, *(const v16h*)(kp + 32), s);
      float* op = &sS[(h * 2 + t) * 256 + l * 8];
      for (int j = 0; j < 8; ++j) op[j] = s[j];
    }
    __syncthreads();
    for (int t = 0; t < 2; ++t) {
      float madd = mask[b * 2048 + m0 + t * 16 + lg] ? 0.f : NEGBIG;
      for (int j = 0; j < 8; ++j) {
        float L = blh + madd;
        for (int g = 0; g < 16; ++g) L += wl[g] * sS[(g * 2 + t) * 256 + l * 8 + j];
        float P = __expf(L - Mrow[j]) * iS[j];
        sP[(h * 16 + j + 8 * lh) * 32 + t * 16 + lg] = (_Float16)P;
      }
    }
    __syncthreads();
    v16h a2h;
    for (int kk = 0; kk < 16; ++kk) {
      float a2 = bwh;
      for (int g = 0; g < 16; ++g)
        a2 += ww[g] * (float)sP[(g * 16 + lg) * 32 + lh * 16 + kk];
      a2h[kk] = (_Float16)a2;
    }
    for (int dt = 0; dt < 4; ++dt) {
      const _Float16* vp = vT + ((b * 16 + h) * 64 + dt * 16 + lg) * 2048 + m0 + lh * 16;
      o[dt] = wmma_f16(a2h, *(const v16h*)vp, o[dt]);
    }
    // next iteration's first barrier orders the sP reuse
  }
  for (int dt = 0; dt < 4; ++dt)
    for (int j = 0; j < 8; ++j) {
      int r = n0 + j + 8 * lh;
      int col = h * 64 + dt * 16 + lg;
      aout[(b * 2048 + r) * 1024 + col] = (_Float16)o[dt][j];
    }
}

// ---------------------------------------------------------------------------
// Kernel 4: output projection.  aout[4096,1024] f16 @ WpT[1024,1024] f16 + bp.
// Same async double-buffered pipeline as kernel 1; f32 result to d_out.
// ---------------------------------------------------------------------------
__global__ __launch_bounds__(256) void proj_gemm(
    const _Float16* __restrict__ A, const _Float16* __restrict__ WT,
    const float* __restrict__ bias, float* __restrict__ out)
{
  __shared__ _Float16 As[2][64 * 32];
  __shared__ _Float16 Bs[2][128 * 32];
  const int tid = threadIdx.x;
  const int l  = tid & 31;
  const int w  = tid >> 5;
  const int wr = w & 3, wc = w >> 2;
  const int lg = l & 15, lh = l >> 4;
  const int row0 = blockIdx.y * 64;
  const int col0 = blockIdx.x * 128;

  const unsigned asB = lds_off_of(&As[0][0]);
  const unsigned bsB = lds_off_of(&Bs[0][0]);
  const int ar = tid >> 2, ap = tid & 3;
  const unsigned aLds = (unsigned)((ar * 32 + ap * 8) * 2);
  const _Float16* aG = A + (row0 + ar) * 1024 + ap * 8;
  const int bc = tid >> 2, bp = tid & 3;
  const unsigned bLds0 = (unsigned)((bc * 32 + bp * 8) * 2);
  const unsigned bLds1 = (unsigned)(((bc + 64) * 32 + bp * 8) * 2);
  const _Float16* bG0 = WT + (col0 + bc) * 1024 + bp * 8;
  const _Float16* bG1 = WT + (col0 + bc + 64) * 1024 + bp * 8;

  async_copy16(asB + aLds, aG);
  async_copy16(bsB + bLds0, bG0);
  async_copy16(bsB + bLds1, bG1);

  v8f acc[4] = {};
  for (int it = 0; it < 32; ++it) {
    const int cur = it & 1;
    if (it + 1 < 32) {
      const int nk = (it + 1) * 32;
      const unsigned ao = asB + (unsigned)((cur ^ 1) * 64 * 32 * 2);
      const unsigned bo = bsB + (unsigned)((cur ^ 1) * 128 * 32 * 2);
      async_copy16(ao + aLds, aG + nk);
      async_copy16(bo + bLds0, bG0 + nk);
      async_copy16(bo + bLds1, bG1 + nk);
      WAIT_ASYNC(0x3);
    } else {
      WAIT_ASYNC(0x0);
    }
    __syncthreads();
    const _Float16* Ac = As[cur];
    const _Float16* Bc = Bs[cur];
    v16h a = *(const v16h*)&Ac[(wr * 16 + lg) * 32 + lh * 16];
    for (int c = 0; c < 4; ++c) {
      v16h bf = *(const v16h*)&Bc[(wc * 64 + c * 16 + lg) * 32 + lh * 16];
      acc[c] = wmma_f16(a, bf, acc[c]);
    }
    __syncthreads();
  }
  for (int c = 0; c < 4; ++c)
    for (int j = 0; j < 8; ++j) {
      int gr = row0 + wr * 16 + j + 8 * lh;
      int gc = col0 + wc * 64 + c * 16 + lg;
      out[gr * 1024 + gc] = acc[c][j] + bias[gc];
    }
}

extern "C" void kernel_launch(void* const* d_in, const int* in_sizes, int n_in,
                              void* d_out, int out_size, void* d_ws, size_t ws_size,
                              hipStream_t stream) {
  const float*         x    = (const float*)d_in[0];
  const unsigned char* mask = (const unsigned char*)d_in[1];   // jnp.bool_
  const float* Wqkv = (const float*)d_in[2];
  const float* bqkv = (const float*)d_in[3];
  const float* Wl   = (const float*)d_in[4];
  const float* bl   = (const float*)d_in[5];
  const float* Ww   = (const float*)d_in[6];
  const float* bw   = (const float*)d_in[7];
  const float* Wp   = (const float*)d_in[8];
  const float* bp   = (const float*)d_in[9];
  float* out = (float*)d_out;

  char* ws = (char*)d_ws;
  size_t off = 0;
  const size_t headElems = (size_t)Bb * Hh * Nn * Dd;             // 4 Mi
  _Float16* qb    = (_Float16*)(ws + off); off += headElems * 2;  // 8 MB
  _Float16* kbuf  = (_Float16*)(ws + off); off += headElems * 2;  // 8 MB
  _Float16* vT    = (_Float16*)(ws + off); off += headElems * 2;  // 8 MB
  float* Mst      = (float*)(ws + off);    off += (size_t)Bb * Hh * Nn * 4;
  float* Sst      = (float*)(ws + off);    off += (size_t)Bb * Hh * Nn * 4;
  _Float16* aout  = (_Float16*)(ws + off); off += (size_t)Bb * Nn * Cc * 2;
  _Float16* xh    = (_Float16*)(ws + off); off += (size_t)Bb * Nn * Cc * 2;
  _Float16* WqkvT = (_Float16*)(ws + off); off += (size_t)Cc * 3 * Cc * 2;
  _Float16* WpT   = (_Float16*)(ws + off); off += (size_t)Cc * Cc * 2;

  const int nx4 = Bb * Nn * Cc / 4;
  cvt_f16<<<(nx4 + 255) / 256, 256, 0, stream>>>(x, xh, nx4);
  transpose_f16<<<dim3(3 * Cc / 32, Cc / 32), 256, 0, stream>>>(Wqkv, WqkvT, Cc, 3 * Cc);
  transpose_f16<<<dim3(Cc / 32, Cc / 32), 256, 0, stream>>>(Wp, WpT, Cc, Cc);

  qkv_gemm<<<dim3(3 * Cc / 128, Bb * Nn / 64), 256, 0, stream>>>(
      xh, WqkvT, bqkv, qb, kbuf, vT);
  attn_stats<<<Bb * (Nn / 16), 512, 0, stream>>>(
      qb, kbuf, mask, Wl, bl, Mst, Sst);
  attn_out_k<<<Bb * (Nn / 16), 512, 0, stream>>>(
      qb, kbuf, vT, mask, Wl, bl, Ww, bw, Mst, Sst, aout);
  proj_gemm<<<dim3(Cc / 128, Bb * Nn / 64), 256, 0, stream>>>(
      aout, WpT, bp, out);
}
